// AttnDecoderRNN_24300924961385
// MI455X (gfx1250) — compile-verified, matrix operand order checked by reading, not run on hardware
//
#include <hip/hip_runtime.h>
#include <hip/hip_bf16.h>
#include <math.h>

// Sizes
#define H  1024
#define V  50257
#define L  2048
#define KA 3072   // attn GEMM K = 2H + H
#define K4 4096   // gate / out GEMV K

#define LDSA_STRIDE 3080   // KA + 8 bf16 pad -> 6160 B/row: per-row bank offset 4, conflict-free frags

typedef __attribute__((ext_vector_type(16))) __bf16 v16bf;
typedef __attribute__((ext_vector_type(8)))  __bf16 v8bf;
typedef __attribute__((ext_vector_type(8)))  float  v8f;

// ---------------- workspace layout (bytes) ----------------
#define OFF_A        ((size_t)0)                 // attn_in bf16  [L x KA]   12,582,912
#define OFF_BT       ((size_t)12582912)          // W_attn^T bf16 [H x KA]    6,291,456
#define OFF_SCORES   ((size_t)18874368)          // f32 [L]
#define OFF_ATTNW    ((size_t)18882560)          // f32 [L]
#define OFF_CONTEXT  ((size_t)18890752)          // f32 [2H]
#define OFF_COMB     ((size_t)18898944)          // f32 [4096]  [emb|context|hidden]
#define OFF_COMB2    ((size_t)18915328)          // f32 [4096]  [emb|hidden|context]
#define OFF_GATES    ((size_t)18931712)          // f32 [4*1024]
#define OFF_LOGITS   ((size_t)18948096)          // f32 [V]
#define OFF_SCAL     ((size_t)19149312)          // f32 [2] : max, log(sum)

// ---------------- bf16 conversion pre-pass ----------------
__global__ __launch_bounds__(256) void cast_attn_in(const float* __restrict__ enc,
                                                    const float* __restrict__ hidden,
                                                    __bf16* __restrict__ A) {
  int idx = blockIdx.x * 256 + threadIdx.x;          // L*KA elements
  int l = idx / KA, c = idx % KA;
  float v = (c < 2 * H) ? enc[(size_t)l * (2 * H) + c] : hidden[c - 2 * H];
  A[idx] = (__bf16)v;
}

__global__ __launch_bounds__(256) void cast_wattnT(const float* __restrict__ W,
                                                   __bf16* __restrict__ BT) {
  int idx = blockIdx.x * 256 + threadIdx.x;          // H*KA elements
  int n = idx / KA, k = idx % KA;
  BT[idx] = (__bf16)W[(size_t)k * H + n];
}

// ---------------- fused attention scores: tanh(A@W + b) @ w_out ----------------
// One block = one 16-row strip of A, staged in LDS via global_load_async_to_lds_b128.
// 8 waves; each wave owns 8 of the 64 N-tiles. B streams from global (L2-resident).
__global__ __launch_bounds__(256) void attn_scores_wmma(
    const __bf16* __restrict__ A,          // [L x KA] bf16
    const __bf16* __restrict__ BT,         // [H x KA] bf16 (W_attn^T)
    const float*  __restrict__ b_attn,     // [H]
    const float*  __restrict__ w_attn_out, // [H]
    const float*  __restrict__ b_attn_out, // [1]
    float* __restrict__ scores)            // [L]
{
  __shared__ __bf16 As[16 * LDSA_STRIDE];  // 16 rows, padded stride (98,560 B)
  __shared__ float  lds_s[16];

  const int m0   = blockIdx.x * 16;
  const int lane = threadIdx.x & 31;
  const int wave = threadIdx.x >> 5;
  const int half = lane >> 4;
  const int r    = lane & 15;

  if (threadIdx.x < 16) lds_s[threadIdx.x] = 0.f;

  // ---- async copy of the A strip into LDS: 16 rows x 6144 B = 6144 b128 xfers ----
  {
    const uint32_t lds_base = (uint32_t)(uintptr_t)(&As[0]);
    for (int idx = threadIdx.x; idx < 16 * (KA / 8); idx += 256) {
      const int row = idx / (KA / 8);          // 0..15
      const int c16 = idx % (KA / 8);          // 16-byte chunk within row
      const uint64_t ga = (uint64_t)(uintptr_t)(A + (size_t)(m0 + row) * KA + c16 * 8);
      const uint32_t la = lds_base + (uint32_t)(row * LDSA_STRIDE + c16 * 8) * 2u;
      asm volatile("global_load_async_to_lds_b128 %0, %1, off"
                   :: "v"(la), "v"(ga) : "memory");
    }
    // each wave drains its own ASYNCcnt, then the workgroup barrier makes all
    // LDS writes visible to every wave
    asm volatile("s_wait_asynccnt 0" ::: "memory");
  }
  __syncthreads();

  float sp[8];
#pragma unroll
  for (int j = 0; j < 8; ++j) sp[j] = 0.f;

  // A-fragment (from LDS): lane holds row r; K chunks at half*8 and half*8+16
  const __bf16* arow = &As[r * LDSA_STRIDE + half * 8];

  for (int nt = wave; nt < 64; nt += 8) {
    const int n0 = nt * 16;
    const __bf16* brow = BT + (size_t)(n0 + r) * KA + half * 8;
    v8f acc = {};
    for (int k0 = 0; k0 < KA; k0 += 32) {
      v8bf a_lo = *(const v8bf*)(arow + k0);        // ds_load_b128
      v8bf a_hi = *(const v8bf*)(arow + k0 + 16);
      v8bf b_lo = *(const v8bf*)(brow + k0);        // global_load_b128
      v8bf b_hi = *(const v8bf*)(brow + k0 + 16);
      v16bf av, bv;
#pragma unroll
      for (int i = 0; i < 8; ++i) {
        av[i] = a_lo[i]; av[8 + i] = a_hi[i];
        bv[i] = b_lo[i]; bv[8 + i] = b_hi[i];
      }
      acc = __builtin_amdgcn_wmma_f32_16x16x32_bf16(
          /*neg_a=*/false, av, /*neg_b=*/false, bv,
          /*c_mod=*/(short)0, acc, /*reuse_a=*/false, /*reuse_b=*/false);
    }
    // epilogue: tanh(acc + b_attn[N]) * w_attn_out[N], accumulated per row M
    const int   N  = n0 + r;
    const float bb = b_attn[N];
    const float wo = w_attn_out[N];
#pragma unroll
    for (int j = 0; j < 8; ++j) sp[j] += tanhf(acc[j] + bb) * wo;
  }

#pragma unroll
  for (int j = 0; j < 8; ++j) atomicAdd(&lds_s[j + half * 8], sp[j]);
  __syncthreads();
  if (threadIdx.x < 16)
    scores[m0 + threadIdx.x] = lds_s[threadIdx.x] + b_attn_out[0];
}

// ---------------- softmax over L=2048 scores (one block) ----------------
__global__ __launch_bounds__(1024) void attn_softmax_kernel(const float* __restrict__ scores,
                                                            float* __restrict__ attn_w,
                                                            float* __restrict__ out_w) {
  __shared__ float red[1024];
  int t = threadIdx.x;
  float s0 = scores[t], s1 = scores[t + 1024];
  red[t] = fmaxf(s0, s1);
  __syncthreads();
  for (int o = 512; o > 0; o >>= 1) { if (t < o) red[t] = fmaxf(red[t], red[t + o]); __syncthreads(); }
  float mx = red[0];
  __syncthreads();
  float e0 = expf(s0 - mx), e1 = expf(s1 - mx);
  red[t] = e0 + e1;
  __syncthreads();
  for (int o = 512; o > 0; o >>= 1) { if (t < o) red[t] += red[t + o]; __syncthreads(); }
  float inv = 1.f / red[0];
  attn_w[t] = e0 * inv;  attn_w[t + 1024] = e1 * inv;
  out_w[t]  = e0 * inv;  out_w[t + 1024]  = e1 * inv;
}

// ---------------- context[j] = sum_l w[l] * enc[l][j] ----------------
__global__ __launch_bounds__(256) void context_kernel(const float* __restrict__ enc,
                                                      const float* __restrict__ attn_w,
                                                      float* __restrict__ context) {
  __shared__ float w[L];
  for (int i = threadIdx.x; i < L; i += 256) w[i] = attn_w[i];
  __syncthreads();
  int j = blockIdx.x * 256 + threadIdx.x;   // 0..2047
  float acc = 0.f;
  for (int l = 0; l < L; ++l) acc = fmaf(w[l], enc[(size_t)l * (2 * H) + j], acc);
  context[j] = acc;
}

// ---------------- build combined vectors (embedding gather fused) ----------------
__global__ __launch_bounds__(256) void build_vectors(const int* __restrict__ tok,
                                                     const float* __restrict__ hidden,
                                                     const float* __restrict__ context,
                                                     const float* __restrict__ W_emb,
                                                     const float* __restrict__ b_emb,
                                                     float* __restrict__ comb,
                                                     float* __restrict__ comb2) {
  int i = blockIdx.x * 256 + threadIdx.x;   // 0..4095
  int t = tok[0];
  float emb_i = (i < H) ? (W_emb[(size_t)t * H + i] + b_emb[i]) : 0.f;
  // comb  = [emb | context | hidden]  (LSTM input)
  float c1 = (i < H) ? emb_i : ((i < 3 * H) ? context[i - H] : hidden[i - 3 * H]);
  // comb2 = [emb | hidden | context]  (output projection input)
  float c2 = (i < H) ? emb_i : ((i < 2 * H) ? hidden[i - H] : context[i - 2 * H]);
  comb[i] = c1;  comb2[i] = c2;
}

// ---------------- 4 gate matvecs (memory-bound, 67 MB) ----------------
__global__ __launch_bounds__(256) void gates_kernel(const float* __restrict__ comb,
    const float* __restrict__ W_i, const float* __restrict__ b_i,
    const float* __restrict__ W_o, const float* __restrict__ b_o,
    const float* __restrict__ W_f, const float* __restrict__ b_f,
    const float* __restrict__ W_c, const float* __restrict__ b_c,
    float* __restrict__ gates) {
  __shared__ float cs[K4];
  for (int i = threadIdx.x; i < K4; i += 256) cs[i] = comb[i];
  __syncthreads();
  const int g = blockIdx.x >> 2;                       // 0..3 : i,o,f,c
  const int n = ((blockIdx.x & 3) * 256) + threadIdx.x;
  const float* Wp = (g == 0) ? W_i : (g == 1) ? W_o : (g == 2) ? W_f : W_c;
  const float* Bp = (g == 0) ? b_i : (g == 1) ? b_o : (g == 2) ? b_f : b_c;
  float acc = Bp[n];
  for (int k = 0; k < K4; ++k) acc = fmaf(cs[k], Wp[(size_t)k * H + n], acc);
  gates[g * H + n] = (g == 3) ? tanhf(acc) : 1.f / (1.f + expf(-acc));
}

// ---------------- LSTM cell update ----------------
__global__ __launch_bounds__(256) void cell_kernel(const float* __restrict__ gates,
                                                   const float* __restrict__ cell,
                                                   float* __restrict__ d_out) {
  int n = blockIdx.x * 256 + threadIdx.x;   // 0..1023
  float i_g = gates[n], o_g = gates[H + n], f_g = gates[2 * H + n], c_g = gates[3 * H + n];
  float nc = f_g * cell[n] + i_g * c_g;
  d_out[V + n] = o_g * tanhf(nc);           // next_hidden
  d_out[V + H + L + n] = nc;                // new_cell
}

// ---------------- output GEMV: 823 MB of W_out, the runtime dominator ----------------
__global__ __launch_bounds__(256) void out_matvec_kernel(const float* __restrict__ c2,
                                                         const float* __restrict__ W_out,
                                                         const float* __restrict__ b_out,
                                                         float* __restrict__ logits) {
  __shared__ float cs[K4];
  for (int i = threadIdx.x; i < K4; i += 256) cs[i] = c2[i];
  __syncthreads();
  int j = blockIdx.x * 256 + threadIdx.x;
  if (j >= V) return;
  const float* col = W_out + j;
  float acc = b_out[j];
  for (int kk = 0; kk < K4; kk += 16) {
    // speculative streaming prefetch 16 rows ahead (global_prefetch_b8)
    __builtin_prefetch(col + (size_t)(kk + 16) * V, 0, 1);
#pragma unroll
    for (int k = kk; k < kk + 16; ++k)
      acc = fmaf(cs[k], col[(size_t)k * V], acc);
  }
  logits[j] = acc;
}

// ---------------- log-sum-exp over V (one block) ----------------
__global__ __launch_bounds__(1024) void lse_kernel(const float* __restrict__ logits,
                                                   float* __restrict__ scal) {
  __shared__ float red[1024];
  int t = threadIdx.x;
  float m = -3.402823e38f;
  for (int j = t; j < V; j += 1024) m = fmaxf(m, logits[j]);
  red[t] = m;
  __syncthreads();
  for (int o = 512; o > 0; o >>= 1) { if (t < o) red[t] = fmaxf(red[t], red[t + o]); __syncthreads(); }
  float mx = red[0];
  __syncthreads();
  float s = 0.f;
  for (int j = t; j < V; j += 1024) s += expf(logits[j] - mx);
  red[t] = s;
  __syncthreads();
  for (int o = 512; o > 0; o >>= 1) { if (t < o) red[t] += red[t + o]; __syncthreads(); }
  if (t == 0) { scal[0] = mx; scal[1] = logf(red[0]); }
}

__global__ __launch_bounds__(256) void write_logsoftmax(const float* __restrict__ logits,
                                                        const float* __restrict__ scal,
                                                        float* __restrict__ d_out) {
  int j = blockIdx.x * 256 + threadIdx.x;
  if (j < V) d_out[j] = logits[j] - scal[0] - scal[1];
}

// ---------------- launcher ----------------
extern "C" void kernel_launch(void* const* d_in, const int* in_sizes, int n_in,
                              void* d_out_v, int out_size, void* d_ws, size_t ws_size,
                              hipStream_t stream) {
  const int*   tok        = (const int*)  d_in[0];
  const float* hidden     = (const float*)d_in[1];
  const float* enc        = (const float*)d_in[2];   // (L,1,2H) contiguous
  const float* cell       = (const float*)d_in[3];
  const float* W_emb      = (const float*)d_in[4];
  const float* b_emb      = (const float*)d_in[5];
  const float* W_attn     = (const float*)d_in[6];
  const float* b_attn     = (const float*)d_in[7];
  const float* W_attn_out = (const float*)d_in[8];
  const float* b_attn_out = (const float*)d_in[9];
  const float* W_i = (const float*)d_in[10]; const float* b_i = (const float*)d_in[11];
  const float* W_o = (const float*)d_in[12]; const float* b_o = (const float*)d_in[13];
  const float* W_f = (const float*)d_in[14]; const float* b_f = (const float*)d_in[15];
  const float* W_c = (const float*)d_in[16]; const float* b_c = (const float*)d_in[17];
  const float* W_out = (const float*)d_in[18]; const float* b_out = (const float*)d_in[19];
  float* d_out = (float*)d_out_v;

  char* ws = (char*)d_ws;   // requires ~19.2 MB of scratch
  __bf16* A       = (__bf16*)(ws + OFF_A);
  __bf16* BT      = (__bf16*)(ws + OFF_BT);
  float*  scores  = (float*)(ws + OFF_SCORES);
  float*  attnw   = (float*)(ws + OFF_ATTNW);
  float*  context = (float*)(ws + OFF_CONTEXT);
  float*  comb    = (float*)(ws + OFF_COMB);
  float*  comb2   = (float*)(ws + OFF_COMB2);
  float*  gates   = (float*)(ws + OFF_GATES);
  float*  logits  = (float*)(ws + OFF_LOGITS);
  float*  scal    = (float*)(ws + OFF_SCAL);

  // 1) bf16 conversions for the WMMA GEMM
  cast_attn_in<<<(L * KA) / 256, 256, 0, stream>>>(enc, hidden, A);
  cast_wattnT<<<(H * KA) / 256, 256, 0, stream>>>(W_attn, BT);

  // 2) fused attention-score GEMM (WMMA, async-LDS staged A) + tanh + projection
  attn_scores_wmma<<<L / 16, 256, 0, stream>>>(A, BT, b_attn, W_attn_out, b_attn_out, scores);

  // 3) softmax -> attn_weights (also final output at d_out[V+H ..])
  attn_softmax_kernel<<<1, 1024, 0, stream>>>(scores, attnw, d_out + V + H);

  // 4) context vector
  context_kernel<<<(2 * H) / 256, 256, 0, stream>>>(enc, attnw, context);

  // 5) combined vectors (embedding gather fused)
  build_vectors<<<K4 / 256, 256, 0, stream>>>(tok, hidden, context, W_emb, b_emb, comb, comb2);

  // 6) LSTM gates + cell update
  gates_kernel<<<16, 256, 0, stream>>>(comb, W_i, b_i, W_o, b_o, W_f, b_f, W_c, b_c, gates);
  cell_kernel<<<H / 256, 256, 0, stream>>>(gates, cell, d_out);

  // 7) output projection over V (memory-bound GEMV, 823 MB of W_out)
  out_matvec_kernel<<<(V + 255) / 256, 256, 0, stream>>>(comb2, W_out, b_out, logits);

  // 8) log_softmax
  lse_kernel<<<1, 1024, 0, stream>>>(logits, scal);
  write_logsoftmax<<<(V + 255) / 256, 256, 0, stream>>>(logits, scal, d_out);
}
